// ptBEVnet_24034636988812
// MI455X (gfx1250) — matrix operand order repaired and checked by reading, not compiled, not run on hardware
//
#include <hip/hip_runtime.h>
#include <stdint.h>

typedef __attribute__((ext_vector_type(16))) _Float16 v16h;
typedef __attribute__((ext_vector_type(8)))  float    v8f;

#define GRID_HW 172800      // G = 480*360
#define NPTS    120000      // points per batch
#define TOTPTS  240000
#define NVOX    345600      // B * G
#define EPSBN   1e-5f

union Frag {
    v16h     h;
    _Float16 f[16];
    uint32_t u[8];
};

// ---------------- workspace layout (bytes) ----------------
#define OFF_W1T   0u          // [64][32]   f16  (4096 B)
#define OFF_W2T   4096u       // [128][64]  f16  (16384 B)
#define OFF_W3T   20480u      // [256][128] f16  (65536 B)
#define OFF_W4T   86016u      // [512][256] f16  (262144 B)
#define OFF_WCT   348160u     // [32][512]  f16  (32768 B)
#define OFF_B1    380928u     // 64  f32
#define OFF_B2    381184u     // 128 f32
#define OFF_B3    381696u     // 256 f32
#define OFF_B4    382720u     // 512 f32
#define OFF_KEYS  385024u     // 345600*512 u32 (order-encoded f32 maxes)
#define NKEYS     (345600ull * 512ull)

// ---------------- prep: fold BN into weights, transpose, f16 ----------------
__global__ void prep_kernel(
    const float* __restrict__ g0, const float* __restrict__ be0,
    const float* __restrict__ m0, const float* __restrict__ v0,
    const float* __restrict__ g1, const float* __restrict__ be1,
    const float* __restrict__ m1, const float* __restrict__ v1,
    const float* __restrict__ g2, const float* __restrict__ be2,
    const float* __restrict__ m2, const float* __restrict__ v2,
    const float* __restrict__ g3, const float* __restrict__ be3,
    const float* __restrict__ m3, const float* __restrict__ v3,
    const float* __restrict__ w1, const float* __restrict__ b1,
    const float* __restrict__ w2, const float* __restrict__ b2,
    const float* __restrict__ w3, const float* __restrict__ b3,
    const float* __restrict__ w4, const float* __restrict__ b4,
    const float* __restrict__ wc,
    _Float16* __restrict__ W1T, _Float16* __restrict__ W2T,
    _Float16* __restrict__ W3T, _Float16* __restrict__ W4T,
    _Float16* __restrict__ WCT,
    float* __restrict__ b1f, float* __restrict__ b2f,
    float* __restrict__ b3f, float* __restrict__ b4f)
{
    int i = blockIdx.x * blockDim.x + threadIdx.x;

    if (i < 64 * 32) {                      // W1T[n][k] = a0[k]*w1[k][n]*a1[n] (k>=7 -> 0)
        int n = i >> 5, k = i & 31;
        float a1 = g1[n] * rsqrtf(v1[n] + EPSBN);
        _Float16 val = (_Float16)0.f;
        if (k < 7) {
            float a0 = g0[k] * rsqrtf(v0[k] + EPSBN);
            val = (_Float16)(a0 * w1[k * 64 + n] * a1);
        }
        W1T[i] = val;
    }
    if (i < 64) {                           // b1f = a1*(b1 + sum c0[k]*w1[k][n]) + c1
        float a1 = g1[i] * rsqrtf(v1[i] + EPSBN);
        float s = b1[i];
        for (int k = 0; k < 7; ++k) {
            float a0 = g0[k] * rsqrtf(v0[k] + EPSBN);
            s += (be0[k] - m0[k] * a0) * w1[k * 64 + i];
        }
        b1f[i] = a1 * s + (be1[i] - m1[i] * a1);
    }
    if (i < 128 * 64) {                     // W2T[n][k] = w2[k][n]*a2[n]
        int n = i >> 6, k = i & 63;
        float a2 = g2[n] * rsqrtf(v2[n] + EPSBN);
        W2T[i] = (_Float16)(w2[k * 128 + n] * a2);
    }
    if (i < 128) {
        float a2 = g2[i] * rsqrtf(v2[i] + EPSBN);
        b2f[i] = a2 * b2[i] + (be2[i] - m2[i] * a2);
    }
    if (i < 256 * 128) {                    // W3T[n][k] = w3[k][n]*a3[n]
        int n = i >> 7, k = i & 127;
        float a3 = g3[n] * rsqrtf(v3[n] + EPSBN);
        W3T[i] = (_Float16)(w3[k * 256 + n] * a3);
    }
    if (i < 256) {
        float a3 = g3[i] * rsqrtf(v3[i] + EPSBN);
        b3f[i] = a3 * b3[i] + (be3[i] - m3[i] * a3);
    }
    if (i < 512 * 256) {                    // W4T[n][k] = w4[k][n]
        int n = i >> 8, k = i & 255;
        W4T[i] = (_Float16)w4[k * 512 + n];
    }
    if (i < 512) b4f[i] = b4[i];
    if (i < 32 * 512) {                     // WCT[n][k] = wc[k][n]
        int n = i >> 9, k = i & 511;
        WCT[i] = (_Float16)wc[k * 32 + n];
    }
}

// ---------------- zero the voxel-key buffer ----------------
__global__ void zero_keys_kernel(uint4* __restrict__ p, long long n4)
{
    long long i = (long long)blockIdx.x * blockDim.x + threadIdx.x;
    long long stride = (long long)gridDim.x * blockDim.x;
    uint4 z; z.x = 0u; z.y = 0u; z.z = 0u; z.w = 0u;
    for (; i < n4; i += stride) p[i] = z;
}

// ---- async global -> LDS panel copy (CDNA5 ASYNCcnt path), then caller barriers ----
__device__ __forceinline__ void panel_copy_async(const _Float16* __restrict__ src,
                                                 _Float16* panel, int bytes, int tid)
{
    uint32_t lds_base = (uint32_t)(uintptr_t)panel;      // low 32 bits = LDS offset
    uint64_t g_base   = (uint64_t)(uintptr_t)src;
    int nchunks = bytes >> 4;                            // 16B per async b128
    for (int i = tid; i < nchunks; i += 256) {
        uint32_t l = lds_base + (uint32_t)(i << 4);
        uint64_t g = g_base + ((uint64_t)i << 4);
        asm volatile("global_load_async_to_lds_b128 %0, %1, off"
                     :: "v"(l), "v"(g) : "memory");
    }
    asm volatile("s_wait_asynccnt 0x0" ::: "memory");    // own loads done; barrier next
}

// ---------------- 2Mx2N register-blocked WMMA layer ----------------
// 8 waves: wm = wave&3 owns rows [wm*32, wm*32+32); wn = wave>>2 owns 32-col half.
// Each round stages a 64-column weight panel; 4 WMMAs per 2A+2B fragments.
template<int KIN, int NOUT, bool RELU>
__device__ __forceinline__ void mlp_layer(
    const _Float16* __restrict__ WT,    // [NOUT][KIN] f16 (L2-resident)
    const float* __restrict__ bias,     // [NOUT] f32
    const _Float16* actIn, int SIN,     // LDS [128][SIN]
    _Float16* actOut, int SOUT,         // LDS [128][SOUT]
    _Float16* panel, int tid)           // LDS [64][KIN]
{
    const int wave = tid >> 5;
    const int lane = tid & 31;
    const int l15  = lane & 15;
    const int hi   = lane >> 4;
    const int wm   = wave & 3;
    const int wn   = wave >> 2;
    const int mb0  = wm * 32;
    const int mb1  = mb0 + 16;

    for (int rnd = 0; rnd < NOUT / 64; ++rnd) {
        __syncthreads();                                    // panel reuse + act visibility
        panel_copy_async(WT + rnd * 64 * KIN, panel, 64 * KIN * 2, tid);
        __syncthreads();

        v8f acc00 = {}, acc01 = {}, acc10 = {}, acc11 = {};
        #pragma unroll
        for (int kt = 0; kt < KIN / 32; ++kt) {
            Frag a0, a1, b0, b1;
            // A: 16x32 f16 ISA layout (lanes 0-15 K{0-7,16-23}, lanes 16-31 K{8-15,24-31})
            #pragma unroll
            for (int v = 0; v < 8; ++v) {
                int kb = (v < 4 ? 2 * v : 16 + 2 * (v - 4)) + 8 * hi;
                a0.u[v] = *(const uint32_t*)(actIn + (mb0 + l15) * SIN + kt * 32 + kb);
                a1.u[v] = *(const uint32_t*)(actIn + (mb1 + l15) * SIN + kt * 32 + kb);
            }
            // B: 32x16 f16, same K layout as A (lanes 0-15 K{0-7,16-23}, lanes 16-31 K{8-15,24-31})
            const uint32_t* p0 = (const uint32_t*)(panel + (wn * 32 + l15) * KIN + kt * 32 + hi * 8);
            const uint32_t* p1 = (const uint32_t*)(panel + (wn * 32 + 16 + l15) * KIN + kt * 32 + hi * 8);
            #pragma unroll
            for (int v = 0; v < 8; ++v) { b0.u[v] = p0[v < 4 ? v : v + 4]; b1.u[v] = p1[v < 4 ? v : v + 4]; }

            acc00 = __builtin_amdgcn_wmma_f32_16x16x32_f16(false, a0.h, false, b0.h, (short)0, acc00, false, false);
            acc01 = __builtin_amdgcn_wmma_f32_16x16x32_f16(false, a0.h, false, b1.h, (short)0, acc01, false, false);
            acc10 = __builtin_amdgcn_wmma_f32_16x16x32_f16(false, a1.h, false, b0.h, (short)0, acc10, false, false);
            acc11 = __builtin_amdgcn_wmma_f32_16x16x32_f16(false, a1.h, false, b1.h, (short)0, acc11, false, false);
        }
        // D layout: VGPR r, lanes 0-15 -> M=r, lanes 16-31 -> M=8+r, N=lane&15
        const int n0 = rnd * 64 + wn * 32 + l15;
        const int n1 = n0 + 16;
        float bv0 = bias[n0], bv1 = bias[n1];
        #pragma unroll
        for (int r = 0; r < 8; ++r) {
            int m0 = mb0 + r + 8 * hi;
            int m1 = mb1 + r + 8 * hi;
            float v00 = acc00[r] + bv0, v01 = acc01[r] + bv1;
            float v10 = acc10[r] + bv0, v11 = acc11[r] + bv1;
            if (RELU) {
                v00 = fmaxf(v00, 0.f); v01 = fmaxf(v01, 0.f);
                v10 = fmaxf(v10, 0.f); v11 = fmaxf(v11, 0.f);
            }
            actOut[m0 * SOUT + n0] = (_Float16)v00;
            actOut[m0 * SOUT + n1] = (_Float16)v01;
            actOut[m1 * SOUT + n0] = (_Float16)v10;
            actOut[m1 * SOUT + n1] = (_Float16)v11;
        }
    }
}

// ---------------- last layer: 2Mx2N WMMA + order-encoded atomic scatter-max ----------------
template<int KIN, int NOUT>
__device__ __forceinline__ void mlp_layer_scatter(
    const _Float16* __restrict__ WT, const float* __restrict__ bias,
    const _Float16* actIn, int SIN,
    _Float16* panel, const int* segs, uint32_t* __restrict__ keys, int tid)
{
    const int wave = tid >> 5;
    const int lane = tid & 31;
    const int l15  = lane & 15;
    const int hi   = lane >> 4;
    const int wm   = wave & 3;
    const int wn   = wave >> 2;
    const int mb0  = wm * 32;
    const int mb1  = mb0 + 16;

    for (int rnd = 0; rnd < NOUT / 64; ++rnd) {
        __syncthreads();
        panel_copy_async(WT + rnd * 64 * KIN, panel, 64 * KIN * 2, tid);
        __syncthreads();

        v8f acc00 = {}, acc01 = {}, acc10 = {}, acc11 = {};
        #pragma unroll
        for (int kt = 0; kt < KIN / 32; ++kt) {
            Frag a0, a1, b0, b1;
            #pragma unroll
            for (int v = 0; v < 8; ++v) {
                int kb = (v < 4 ? 2 * v : 16 + 2 * (v - 4)) + 8 * hi;
                a0.u[v] = *(const uint32_t*)(actIn + (mb0 + l15) * SIN + kt * 32 + kb);
                a1.u[v] = *(const uint32_t*)(actIn + (mb1 + l15) * SIN + kt * 32 + kb);
            }
            const uint32_t* p0 = (const uint32_t*)(panel + (wn * 32 + l15) * KIN + kt * 32 + hi * 8);
            const uint32_t* p1 = (const uint32_t*)(panel + (wn * 32 + 16 + l15) * KIN + kt * 32 + hi * 8);
            #pragma unroll
            for (int v = 0; v < 8; ++v) { b0.u[v] = p0[v < 4 ? v : v + 4]; b1.u[v] = p1[v < 4 ? v : v + 4]; }

            acc00 = __builtin_amdgcn_wmma_f32_16x16x32_f16(false, a0.h, false, b0.h, (short)0, acc00, false, false);
            acc01 = __builtin_amdgcn_wmma_f32_16x16x32_f16(false, a0.h, false, b1.h, (short)0, acc01, false, false);
            acc10 = __builtin_amdgcn_wmma_f32_16x16x32_f16(false, a1.h, false, b0.h, (short)0, acc10, false, false);
            acc11 = __builtin_amdgcn_wmma_f32_16x16x32_f16(false, a1.h, false, b1.h, (short)0, acc11, false, false);
        }
        const int f0 = rnd * 64 + wn * 32 + l15;
        const int f1 = f0 + 16;
        float bv0 = bias[f0], bv1 = bias[f1];
        #pragma unroll
        for (int r = 0; r < 8; ++r) {
            int m0 = mb0 + r + 8 * hi;
            int m1 = mb1 + r + 8 * hi;
            unsigned long long s0 = (unsigned long long)segs[m0] * 512ull;
            unsigned long long s1 = (unsigned long long)segs[m1] * 512ull;
            float v00 = acc00[r] + bv0, v01 = acc01[r] + bv1;
            float v10 = acc10[r] + bv0, v11 = acc11[r] + bv1;
            uint32_t u00 = __float_as_uint(v00), u01 = __float_as_uint(v01);
            uint32_t u10 = __float_as_uint(v10), u11 = __float_as_uint(v11);
            // order-preserving encode; 0 reserved for "empty"
            u00 ^= (u00 & 0x80000000u) ? 0xFFFFFFFFu : 0x80000000u;
            u01 ^= (u01 & 0x80000000u) ? 0xFFFFFFFFu : 0x80000000u;
            u10 ^= (u10 & 0x80000000u) ? 0xFFFFFFFFu : 0x80000000u;
            u11 ^= (u11 & 0x80000000u) ? 0xFFFFFFFFu : 0x80000000u;
            atomicMax(&keys[s0 + (unsigned)f0], u00);
            atomicMax(&keys[s0 + (unsigned)f1], u01);
            atomicMax(&keys[s1 + (unsigned)f0], u10);
            atomicMax(&keys[s1 + (unsigned)f1], u11);
        }
    }
}

// ---------------- fused point MLP ----------------
__global__ __launch_bounds__(256, 2) void mlp_kernel(
    const float* __restrict__ pt_fea, const int* __restrict__ xy_ind,
    const _Float16* __restrict__ W1T, const float* __restrict__ b1f,
    const _Float16* __restrict__ W2T, const float* __restrict__ b2f,
    const _Float16* __restrict__ W3T, const float* __restrict__ b3f,
    const _Float16* __restrict__ W4T, const float* __restrict__ b4f,
    uint32_t* __restrict__ keys)
{
    __shared__ __align__(16) _Float16 actA[128 * 144];   // x0[.,32] / x2[.,128]
    __shared__ __align__(16) _Float16 actB[128 * 272];   // x1[.,64] / x3[.,256]
    __shared__ __align__(16) _Float16 panel[64 * 256];   // 64-col weight panel (max KIN=256)
    __shared__ int segs[128];

    const int tid = threadIdx.x;
    const long long pbase = (long long)blockIdx.x * 128;

    if (tid < 128) {
        long long p = pbase + tid;
        int batch = (int)(p / NPTS);
        segs[tid] = xy_ind[p] + batch * GRID_HW;
    }
    // x0: raw features cast to f16, K padded 7 -> 32 with zeros (BN0 folded into W1T)
    for (int i = tid; i < 128 * 32; i += 256) {
        int r = i >> 5, c = i & 31;
        float v = (c < 7) ? pt_fea[(pbase + r) * 7 + c] : 0.f;
        actA[r * 144 + c] = (_Float16)v;
    }
    // layer-top barriers cover the init stores
    mlp_layer<32, 64, true >(W1T, b1f, actA, 144, actB, 272, panel, tid);
    mlp_layer<64, 128, true>(W2T, b2f, actB, 272, actA, 144, panel, tid);
    mlp_layer<128, 256, true>(W3T, b3f, actA, 144, actB, 272, panel, tid);
    mlp_layer_scatter<256, 512>(W4T, b4f, actB, 272, panel, segs, keys, tid);
}

// ---------------- compression: 512 -> 32 WMMA over voxels, +transpose ----------------
__device__ __forceinline__ float dec_key(uint32_t key)
{
    uint32_t bits = key ^ ((key & 0x80000000u) ? 0x80000000u : 0xFFFFFFFFu);
    return (key == 0u) ? 0.f : __uint_as_float(bits);
}

__global__ __launch_bounds__(256) void comp_kernel(
    const uint32_t* __restrict__ keys,
    const _Float16* __restrict__ WCT,   // [32][512] f16
    const float* __restrict__ bc,       // [32]
    float* __restrict__ out)            // [B,32,480,360]
{
    const int tid  = threadIdx.x;
    const int wave = tid >> 5;
    const int lane = tid & 31;
    const int l15  = lane & 15;
    const int hi   = lane >> 4;
    const long long vb = ((long long)blockIdx.x * 8 + wave) * 16;   // 16 voxels / wave

    v8f acc0 = {}, acc1 = {};
    const long long rowbase = (vb + l15) * 512;
    for (int kt = 0; kt < 16; ++kt) {           // K = 512
        Frag a, b0, b1;
        #pragma unroll
        for (int v = 0; v < 8; ++v) {
            int kb = (v < 4 ? 2 * v : 16 + 2 * (v - 4)) + 8 * hi;
            int k  = kt * 32 + kb;
            a.f[2 * v]     = (_Float16)dec_key(keys[rowbase + k]);
            a.f[2 * v + 1] = (_Float16)dec_key(keys[rowbase + k + 1]);
        }
        const uint32_t* p0 = (const uint32_t*)(WCT + l15 * 512 + kt * 32 + hi * 8);
        const uint32_t* p1 = (const uint32_t*)(WCT + (16 + l15) * 512 + kt * 32 + hi * 8);
        #pragma unroll
        for (int v = 0; v < 8; ++v) { b0.u[v] = p0[v < 4 ? v : v + 4]; b1.u[v] = p1[v < 4 ? v : v + 4]; }

        acc0 = __builtin_amdgcn_wmma_f32_16x16x32_f16(false, a.h, false, b0.h,
                                                      (short)0, acc0, false, false);
        acc1 = __builtin_amdgcn_wmma_f32_16x16x32_f16(false, a.h, false, b1.h,
                                                      (short)0, acc1, false, false);
    }
    float bc0 = bc[l15], bc1 = bc[16 + l15];
    #pragma unroll
    for (int r = 0; r < 8; ++r) {
        int m = r + 8 * hi;
        long long s = vb + m;
        bool occ = keys[s * 512] != 0u;
        int b = (int)(s / GRID_HW);
        int g = (int)(s - (long long)b * GRID_HW);
        float o0 = occ ? fmaxf(acc0[r] + bc0, 0.f) : 0.f;
        float o1 = occ ? fmaxf(acc1[r] + bc1, 0.f) : 0.f;
        out[((long long)b * 32 + l15)      * GRID_HW + g] = o0;
        out[((long long)b * 32 + 16 + l15) * GRID_HW + g] = o1;
    }
}

// ---------------- launch ----------------
extern "C" void kernel_launch(void* const* d_in, const int* in_sizes, int n_in,
                              void* d_out, int out_size, void* d_ws, size_t ws_size,
                              hipStream_t stream)
{
    (void)in_sizes; (void)n_in; (void)out_size; (void)ws_size;

    const float* pt_fea = (const float*)d_in[0];
    const int*   xy_ind = (const int*)d_in[1];
    const float* g0  = (const float*)d_in[2];
    const float* be0 = (const float*)d_in[3];
    const float* m0  = (const float*)d_in[4];
    const float* v0  = (const float*)d_in[5];
    const float* g1  = (const float*)d_in[6];
    const float* be1 = (const float*)d_in[7];
    const float* m1  = (const float*)d_in[8];
    const float* v1  = (const float*)d_in[9];
    const float* g2  = (const float*)d_in[10];
    const float* be2 = (const float*)d_in[11];
    const float* m2  = (const float*)d_in[12];
    const float* v2  = (const float*)d_in[13];
    const float* g3  = (const float*)d_in[14];
    const float* be3 = (const float*)d_in[15];
    const float* m3  = (const float*)d_in[16];
    const float* v3  = (const float*)d_in[17];
    const float* w1  = (const float*)d_in[18];
    const float* b1  = (const float*)d_in[19];
    const float* w2  = (const float*)d_in[20];
    const float* b2  = (const float*)d_in[21];
    const float* w3  = (const float*)d_in[22];
    const float* b3  = (const float*)d_in[23];
    const float* w4  = (const float*)d_in[24];
    const float* b4  = (const float*)d_in[25];
    const float* wc  = (const float*)d_in[26];
    const float* bc  = (const float*)d_in[27];

    char* ws = (char*)d_ws;
    _Float16* W1T = (_Float16*)(ws + OFF_W1T);
    _Float16* W2T = (_Float16*)(ws + OFF_W2T);
    _Float16* W3T = (_Float16*)(ws + OFF_W3T);
    _Float16* W4T = (_Float16*)(ws + OFF_W4T);
    _Float16* WCT = (_Float16*)(ws + OFF_WCT);
    float* b1f = (float*)(ws + OFF_B1);
    float* b2f = (float*)(ws + OFF_B2);
    float* b3f = (float*)(ws + OFF_B3);
    float* b4f = (float*)(ws + OFF_B4);
    uint32_t* keys = (uint32_t*)(ws + OFF_KEYS);

    // 1) clear voxel-key buffer (345600*512 u32)
    zero_keys_kernel<<<8192, 256, 0, stream>>>((uint4*)keys, (long long)(NKEYS / 4));

    // 2) fold BN + transpose weights to f16
    prep_kernel<<<(512 * 256) / 256, 256, 0, stream>>>(
        g0, be0, m0, v0, g1, be1, m1, v1, g2, be2, m2, v2, g3, be3, m3, v3,
        w1, b1, w2, b2, w3, b3, w4, b4, wc,
        W1T, W2T, W3T, W4T, WCT, b1f, b2f, b3f, b4f);

    // 3) fused point MLP + atomic scatter-max (240000 pts / 128 per block)
    mlp_kernel<<<TOTPTS / 128, 256, 0, stream>>>(
        pt_fea, xy_ind, W1T, b1f, W2T, b2f, W3T, b3f, W4T, b4f, keys);

    // 4) per-voxel compression GEMM + ReLU + transpose (345600 vox / 128 per block)
    comp_kernel<<<NVOX / 128, 256, 0, stream>>>(keys, WCT, bc, (float*)d_out);
}
